// MultiHeadedAttention_48954037240254
// MI455X (gfx1250) — matrix operand order repaired and checked by reading, not compiled, not run on hardware
//
#include <hip/hip_runtime.h>

typedef _Float16 half_t;
typedef __attribute__((ext_vector_type(16))) _Float16 v16h;
typedef __attribute__((ext_vector_type(8)))  _Float16 v8h;
typedef __attribute__((ext_vector_type(4)))  _Float16 v4h;
typedef __attribute__((ext_vector_type(8)))  float    v8f;
typedef __attribute__((ext_vector_type(4)))  unsigned int v4u;
typedef __attribute__((ext_vector_type(8)))  int      v8i_;
typedef __attribute__((ext_vector_type(4)))  int      v4i_;

#if defined(__has_builtin)
#if __has_builtin(__builtin_amdgcn_tensor_load_to_lds)
#define HAVE_TDM 1
#endif
#endif
#ifndef HAVE_TDM
#define HAVE_TDM 0
#endif

#define WMMA_F32_F16(a,b,c) \
  __builtin_amdgcn_wmma_f32_16x16x32_f16(false,(a),false,(b),(short)0,(c),false,false)

__device__ __forceinline__ v16h combine16(v8h lo, v8h hi) {
  return __builtin_shufflevector(lo, hi, 0,1,2,3,4,5,6,7,8,9,10,11,12,13,14,15);
}

// ---------------------------------------------------------------------------
// TDM: load a 2D tile of f16 (tile_d0 elements per row, tile_d1 rows, global
// row stride stride0 elements) into LDS with hardware row padding.
// D# bitfields per CDNA5 ISA. Wave-level op (EXEC ignored).
// pad codes: interval -> 2^(c+1) DWORDs between pads, amount -> (c+1) DWORDs.
// ---------------------------------------------------------------------------
__device__ __forceinline__ void tdm_load_2d(unsigned lds_addr, const half_t* gptr,
                                            unsigned tile_d0, unsigned tile_d1,
                                            unsigned long stride0,
                                            unsigned pad_int_c, unsigned pad_amt_c)
{
#if HAVE_TDM
  unsigned long ga = (unsigned long)gptr;
  v4u g0;
  g0[0] = 1u;                                            // count=1, user mode
  g0[1] = lds_addr;                                      // lds_addr [63:32]
  g0[2] = (unsigned)(ga & 0xFFFFFFFFul);                 // global_addr lo
  g0[3] = (unsigned)((ga >> 32) & 0x01FFFFFFul) | (2u << 30);  // ga hi | type=2
  v8i_ g1;
  g1[0] = (int)((1u << 16)                               // data_size = 2 bytes
              | (1u << 20)                               // pad_enable
              | (pad_int_c << 22) | (pad_amt_c << 25));
  g1[1] = (int)((tile_d0 & 0xFFFFu) << 16);              // tensor_dim0 lo16
  g1[2] = (int)(((tile_d0 >> 16) & 0xFFFFu) | ((tile_d1 & 0xFFFFu) << 16));
  g1[3] = (int)(((tile_d1 >> 16) & 0xFFFFu) | ((tile_d0 & 0xFFFFu) << 16)); // tile_dim0
  g1[4] = (int)(tile_d1 & 0xFFFFu);                      // tile_dim1, tile_dim2=0
  g1[5] = (int)(unsigned)(stride0 & 0xFFFFFFFFul);       // tensor_dim0_stride lo32
  g1[6] = (int)(unsigned)((stride0 >> 32) & 0xFFFFu);    // stride hi16
  g1[7] = 0;
  v4i_ gz = {};
#if __clang_major__ >= 23
  v8i_ gz8 = {};
  __builtin_amdgcn_tensor_load_to_lds(g0, g1, gz, gz, gz8, 0);
#else
  __builtin_amdgcn_tensor_load_to_lds(g0, g1, gz, gz, 0);
#endif
#else
  (void)lds_addr; (void)gptr; (void)tile_d0; (void)tile_d1;
  (void)stride0; (void)pad_int_c; (void)pad_amt_c;
#endif
}

__device__ __forceinline__ unsigned lds_lo32(const void* p) {
  return (unsigned)(unsigned long)p;   // generic LDS ptr: low 32 bits = LDS addr
}

// ---------------------------------------------------------------------------
// Pre-passes: f32 -> f16 convert, and W[K,N] f32 -> Wt[N,K] f16 transpose.
// ---------------------------------------------------------------------------
__global__ __launch_bounds__(256)
void convert_f16_kernel(const float* __restrict__ in, half_t* __restrict__ out, int n4)
{
  int i = blockIdx.x * 256 + threadIdx.x;
  if (i < n4) {
    float4 f = *(const float4*)(in + (size_t)i * 4);
    v4h h = { (half_t)f.x, (half_t)f.y, (half_t)f.z, (half_t)f.w };
    *(v4h*)(out + (size_t)i * 4) = h;
  }
}

__global__ __launch_bounds__(256)
void transpose_f16_kernel(const float* __restrict__ W, half_t* __restrict__ Wt,
                          int K, int N)
{
  __shared__ half_t t[32][33];
  const int k0 = blockIdx.y * 32, n0 = blockIdx.x * 32;
  const int tx = threadIdx.x & 31, ty = threadIdx.x >> 5;   // 32 x 8
  #pragma unroll
  for (int r = ty; r < 32; r += 8)
    t[r][tx] = (half_t)W[(size_t)(k0 + r) * N + n0 + tx];
  __syncthreads();
  #pragma unroll
  for (int r = ty; r < 32; r += 8)
    Wt[(size_t)(n0 + r) * K + k0 + tx] = t[tx][r];
}

// ---------------------------------------------------------------------------
// C[M,N] (OT) = A[M,K] (f16) @ Bt[N,K]^T (f16) + bias (f32)
// 256 threads = 8 waves, block tile 128x128, wave tile 32x64, K-step 32.
// TDM double-buffered tile staging (wave 0 prefetches tile k+1 during tile k).
// ---------------------------------------------------------------------------
template<typename OT>
__global__ __launch_bounds__(256)
void gemm_bias_wmma(const half_t* __restrict__ A, const half_t* __restrict__ Bw,
                    const float* __restrict__ bias, OT* __restrict__ C,
                    int M, int N, int K)
{
  constexpr int LDT = 40;                       // padded halves per 32-k row
  __shared__ half_t Als[2][128 * LDT];
  __shared__ half_t Bls[2][128 * LDT];

  const int tid  = threadIdx.x;
  const int lane = tid & 31, wid = tid >> 5;
  const int g = lane >> 4, l16 = lane & 15;
  const int wM = wid >> 1, wN = wid & 1;
  const int m0 = blockIdx.y * 128, n0 = blockIdx.x * 128;
  const half_t* Ag = A  + (size_t)m0 * K;       // 128 x K panel
  const half_t* Bg = Bw + (size_t)n0 * K;       // 128 x K panel (Bt rows = n)

  v8f acc[2][4] = {};
  const int nk = K >> 5;

#if HAVE_TDM
  if (wid == 0) {                               // prologue: tile 0 -> buf 0
    tdm_load_2d(lds_lo32(&Als[0][0]), Ag, 32, 128, (unsigned long)K, 3, 3);
    tdm_load_2d(lds_lo32(&Bls[0][0]), Bg, 32, 128, (unsigned long)K, 3, 3);
  }
#endif

  for (int kt = 0; kt < nk; kt++) {
    const int buf = kt & 1;
#if HAVE_TDM
    if (wid == 0) {
      if (kt + 1 < nk) {
        tdm_load_2d(lds_lo32(&Als[buf ^ 1][0]), Ag + (kt + 1) * 32, 32, 128,
                    (unsigned long)K, 3, 3);
        tdm_load_2d(lds_lo32(&Bls[buf ^ 1][0]), Bg + (kt + 1) * 32, 32, 128,
                    (unsigned long)K, 3, 3);
        __builtin_amdgcn_s_wait_tensorcnt(2);   // tile kt complete (in-order)
      } else {
        __builtin_amdgcn_s_wait_tensorcnt(0);
      }
    }
    __syncthreads();
#else
    __syncthreads();
    #pragma unroll
    for (int i = 0; i < 2; i++) {               // 512 x 8-half chunks, 2/thread
      int e = tid + 256 * i;
      int r = e >> 2, cs = (e & 3) * 8;
      *(v8h*)&Als[buf][r * LDT + cs] = *(const v8h*)&Ag[(size_t)r * K + kt * 32 + cs];
      *(v8h*)&Bls[buf][r * LDT + cs] = *(const v8h*)&Bg[(size_t)r * K + kt * 32 + cs];
    }
    __syncthreads();
#endif

    // A fragments: lane l16 = row, 16B chunks at k=8g and k=16+8g (ISA layout)
    v16h af[2];
    #pragma unroll
    for (int mt = 0; mt < 2; mt++) {
      const half_t* p = &Als[buf][(wM * 32 + mt * 16 + l16) * LDT];
      af[mt] = combine16(*(const v8h*)(p + 8 * g), *(const v8h*)(p + 16 + 8 * g));
    }
    // B fragments: lane l16 = col n, 16B chunks at k=8g and k=16+8g of Bt row
    #pragma unroll
    for (int nt = 0; nt < 4; nt++) {
      const half_t* p = &Bls[buf][(wN * 64 + nt * 16 + l16) * LDT];
      v16h bfrag = combine16(*(const v8h*)(p + 8 * g), *(const v8h*)(p + 16 + 8 * g));
      #pragma unroll
      for (int mt = 0; mt < 2; mt++)
        acc[mt][nt] = WMMA_F32_F16(af[mt], bfrag, acc[mt][nt]);
    }
    __syncthreads();
  }

  // epilogue: C/D layout row = v + 8g, col = l16
  #pragma unroll
  for (int mt = 0; mt < 2; mt++)
    #pragma unroll
    for (int nt = 0; nt < 4; nt++) {
      int col = n0 + wN * 64 + nt * 16 + l16;
      float bv = bias[col];
      #pragma unroll
      for (int v = 0; v < 8; v++) {
        int row = m0 + wM * 32 + mt * 16 + v + 8 * g;
        C[(size_t)row * N + col] = (OT)(acc[mt][nt][v] + bv);
      }
    }
}

// ---------------------------------------------------------------------------
// Causal flash attention. Grid (S/64, H, B), 128 threads = 4 waves,
// each wave owns 16 query rows. K tile TDM-loaded; V transposed manually.
// ---------------------------------------------------------------------------
__global__ __launch_bounds__(128)
void attn_fwd_wmma(const half_t* __restrict__ Qm, const half_t* __restrict__ KVm,
                   half_t* __restrict__ Om)
{
  constexpr int S = 2048, D = 1024, KVLD = 2048, DK = 64, LDT = 72;
  __shared__ half_t Kl[64 * LDT];          // [kn][d]  (B layout for QK^T)
  __shared__ half_t Vt[64 * LDT];          // [d][kn]  (B layout for PV)
  __shared__ half_t Pl[4 * 16 * LDT];      // per-wave P staging [16][64]

  const int tid  = threadIdx.x;
  const int lane = tid & 31, wid = tid >> 5;
  const int g = lane >> 4, l16 = lane & 15;
  const int qb = blockIdx.x, h = blockIdx.y, b = blockIdx.z;
  const int qBase = qb * 64;
  const int qRow0 = qBase + wid * 16;

  // Q fragments live in registers across the whole KV loop
  const half_t* qp = Qm + (size_t)(b * S + qRow0 + l16) * D + h * DK;
  v16h qf[2];
  #pragma unroll
  for (int c = 0; c < 2; c++)
    qf[c] = combine16(*(const v8h*)(qp + 32 * c + 8 * g),
                      *(const v8h*)(qp + 32 * c + 16 + 8 * g));

  v8f o[4] = {};
  float mrun[8], lrun[8];
  #pragma unroll
  for (int v = 0; v < 8; v++) { mrun[v] = -3.0e38f; lrun[v] = 0.0f; }

  for (int kvb = 0; kvb <= qb; kvb++) {     // causal: blocks on/below diagonal
    const int kvBase = kvb * 64;
    const half_t* kg = KVm + (size_t)(b * S + kvBase) * KVLD + h * DK;
    __syncthreads();
#if HAVE_TDM
    if (wid == 0)
      tdm_load_2d(lds_lo32(&Kl[0]), kg, 64, 64, (unsigned long)KVLD, 4, 3);
#else
    for (int e = tid; e < 64 * 64; e += 128) {
      int kn = e >> 6, d = e & 63;
      Kl[kn * LDT + d] = kg[(size_t)kn * KVLD + d];
    }
#endif
    for (int e = tid; e < 64 * 64; e += 128) {         // V transpose (manual)
      int kn = e >> 6, d = e & 63;
      Vt[d * LDT + kn] = kg[(size_t)kn * KVLD + D + d];
    }
#if HAVE_TDM
    if (wid == 0) __builtin_amdgcn_s_wait_tensorcnt(0);
#endif
    __syncthreads();

    // scores = Q @ K^T  (4 tiles of 16 keys, 2 k-chunks over DK=64)
    v8f s[4] = {};
    #pragma unroll
    for (int nt = 0; nt < 4; nt++) {
      #pragma unroll
      for (int c = 0; c < 2; c++) {
        const half_t* p = &Kl[(nt * 16 + l16) * LDT + 32 * c];
        v16h bk = combine16(*(const v8h*)(p + 8 * g), *(const v8h*)(p + 16 + 8 * g));
        s[nt] = WMMA_F32_F16(qf[c], bk, s[nt]);
      }
    }

    const bool diag = (kvb == qb);
    #pragma unroll
    for (int nt = 0; nt < 4; nt++)
      #pragma unroll
      for (int v = 0; v < 8; v++) {
        float sv = s[nt][v] * 0.125f;       // 1/sqrt(64)
        if (diag) {
          int qm = qRow0 + v + 8 * g;
          int kn = kvBase + nt * 16 + l16;
          if (kn > qm) sv = -1.0e9f;
        }
        s[nt][v] = sv;
      }

    // online softmax; row of element v is (v + 8g); 16-lane-half reductions
    #pragma unroll
    for (int v = 0; v < 8; v++) {
      float tmax = fmaxf(fmaxf(s[0][v], s[1][v]), fmaxf(s[2][v], s[3][v]));
      #pragma unroll
      for (int msk = 1; msk < 16; msk <<= 1)
        tmax = fmaxf(tmax, __shfl_xor(tmax, msk, 32));
      float mnew = fmaxf(mrun[v], tmax);
      float corr = __expf(mrun[v] - mnew);
      mrun[v] = mnew;
      float rsum = 0.0f;
      #pragma unroll
      for (int nt = 0; nt < 4; nt++) {
        float pv = __expf(s[nt][v] - mnew);
        s[nt][v] = pv;
        rsum += pv;
      }
      #pragma unroll
      for (int msk = 1; msk < 16; msk <<= 1)
        rsum += __shfl_xor(rsum, msk, 32);
      lrun[v] = lrun[v] * corr + rsum;
      #pragma unroll
      for (int dt = 0; dt < 4; dt++) o[dt][v] *= corr;
      #pragma unroll
      for (int nt = 0; nt < 4; nt++)
        Pl[(wid * 16 + v + 8 * g) * LDT + nt * 16 + l16] = (half_t)s[nt][v];
    }

    // O += P @ V   (per-wave LDS round-trip converts C-layout -> A-layout)
    #pragma unroll
    for (int c = 0; c < 2; c++) {
      const half_t* pp = &Pl[(wid * 16 + l16) * LDT + 32 * c];
      v16h pf = combine16(*(const v8h*)(pp + 8 * g), *(const v8h*)(pp + 16 + 8 * g));
      #pragma unroll
      for (int dt = 0; dt < 4; dt++) {
        const half_t* vp = &Vt[(dt * 16 + l16) * LDT + 32 * c];
        v16h bv = combine16(*(const v8h*)(vp + 8 * g), *(const v8h*)(vp + 16 + 8 * g));
        o[dt] = WMMA_F32_F16(pf, bv, o[dt]);
      }
    }
  }

  // normalize and store O as f16 for the final projection
  #pragma unroll
  for (int v = 0; v < 8; v++) {
    float inv = 1.0f / lrun[v];
    int row = qRow0 + v + 8 * g;
    #pragma unroll
    for (int dt = 0; dt < 4; dt++)
      Om[(size_t)(b * S + row) * D + h * DK + dt * 16 + l16] =
          (half_t)(o[dt][v] * inv);
  }
}

// ---------------------------------------------------------------------------
extern "C" void kernel_launch(void* const* d_in, const int* in_sizes, int n_in,
                              void* d_out, int out_size, void* d_ws, size_t ws_size,
                              hipStream_t stream)
{
  const float* x   = (const float*)d_in[0];
  // d_in[1] = mask: causal tril, implemented analytically in the kernel
  const float* Wq  = (const float*)d_in[2];
  const float* bq  = (const float*)d_in[3];
  const float* Wkv = (const float*)d_in[4];
  const float* bkv = (const float*)d_in[5];
  const float* Wf  = (const float*)d_in[6];
  const float* bf  = (const float*)d_in[7];
  float* out = (float*)d_out;

  const int M = 4 * 2048;                 // B*S
  const int D = 1024;
  const size_t MiB = 1024 * 1024;

  char* ws = (char*)d_ws;
  half_t* x_h   = (half_t*)(ws);                    // 16 MiB
  half_t* q_h   = (half_t*)(ws + 16 * MiB);         // 16 MiB
  half_t* kv_h  = (half_t*)(ws + 32 * MiB);         // 32 MiB
  half_t* o_h   = (half_t*)(ws + 64 * MiB);         // 16 MiB
  half_t* wt_q  = (half_t*)(ws + 80 * MiB);         //  2 MiB  [N=1024][K=1024]
  half_t* wt_kv = (half_t*)(ws + 82 * MiB);         //  4 MiB  [N=2048][K=1024]
  half_t* wt_f  = (half_t*)(ws + 86 * MiB);         //  2 MiB

  // pre-passes: convert x, transpose+convert weights (one-time, ~25MB traffic)
  {
    int n4 = (M * D) / 4;
    convert_f16_kernel<<<dim3((n4 + 255) / 256), dim3(256), 0, stream>>>(x, x_h, n4);
    transpose_f16_kernel<<<dim3(D / 32, D / 32), dim3(256), 0, stream>>>(Wq, wt_q, D, D);
    transpose_f16_kernel<<<dim3(2 * D / 32, D / 32), dim3(256), 0, stream>>>(Wkv, wt_kv, D, 2 * D);
    transpose_f16_kernel<<<dim3(D / 32, D / 32), dim3(256), 0, stream>>>(Wf, wt_f, D, D);
  }

  dim3 blk(256);
  gemm_bias_wmma<half_t>
      <<<dim3(D / 128, M / 128), blk, 0, stream>>>(x_h, wt_q, bq, q_h, M, D, D);
  gemm_bias_wmma<half_t>
      <<<dim3(2 * D / 128, M / 128), blk, 0, stream>>>(x_h, wt_kv, bkv, kv_h, M, 2 * D, D);
  attn_fwd_wmma<<<dim3(32, 16, 4), dim3(128), 0, stream>>>(q_h, kv_h, o_h);
  gemm_bias_wmma<float>
      <<<dim3(D / 128, M / 128), blk, 0, stream>>>(o_h, wt_f, bf, out, M, D, D);
}
